// MultiHeadedAttentionWithRelations_36593121362377
// MI455X (gfx1250) — matrix operand order repaired and checked by reading, not compiled, not run on hardware
//
#include <hip/hip_runtime.h>
#include <hip/hip_bf16.h>

// MultiHeadedAttentionWithRelations for MI455X (gfx1250, wave32, WMMA).
// Memory-bound on relation_k/relation_v (537 MB f32); all matmuls use
// v_wmma_f32_16x16x32_bf16 with f32 accumulation. bf16 tile staging uses the
// CDNA5 async global->LDS path (ASYNCcnt); f32 tiles convert via VALU.

#define Bz  4
#define Sz  512
#define Hz  1024
#define NHz 16
#define HDz 64

typedef __attribute__((ext_vector_type(16))) __bf16 v16bf;
typedef __attribute__((ext_vector_type(8)))  float  v8f;

static __device__ __forceinline__ int laneId() { return threadIdx.x & 31; }
static __device__ __forceinline__ int waveId() { return threadIdx.x >> 5; }

static __device__ __forceinline__ v8f wmma_bf16(v16bf a, v16bf b, v8f c) {
  // 8 args: (neg_a, A, neg_b, B, c_mod, C, reuse_a, reuse_b)
  return __builtin_amdgcn_wmma_f32_16x16x32_bf16(false, a, false, b, (short)0, c,
                                                 false, false);
}

// ---- CDNA5 async global->LDS copy (16 bytes / lane, ASYNCcnt tracked) ------
// Generic pointers to __shared__ memory carry the LDS byte offset in their low
// 32 bits.
static __device__ __forceinline__ void asyncCopy16(const void* gsrc, void* ldst) {
  const unsigned d = (unsigned)(size_t)ldst;
  asm volatile("global_load_async_to_lds_b128 %0, %1, off"
               :: "v"(d), "v"(gsrc) : "memory");
}
static __device__ __forceinline__ void asyncWait0() {
  asm volatile("s_wait_asynccnt 0" ::: "memory");
}

// ---- WMMA fragment gathers from LDS (contiguous k = 16*(lane/16) + s in both A and B) -------------------

// A fragment: 16x32 tile, row-major in LDS, row stride `ld` elements.
static __device__ __forceinline__ v16bf ldsLoadA(const __bf16* base, int ld) {
  const int l = laneId();
  const int m = l & 15;
  const int kh = (l >> 4) << 4;
  union { v16bf v; unsigned u[8]; } a;
  const __bf16* row = base + m * ld + kh;
#pragma unroll
  for (int g = 0; g < 8; ++g) {
    const int k = g << 1;
    a.u[g] = *(const unsigned*)(row + k);   // packed bf16 pair (4B aligned)
  }
  return a.v;
}

// B fragment: 32x16 (K x N) tile stored N-major in LDS: elem(k,n)=base[n*ld+k].
static __device__ __forceinline__ v16bf ldsLoadB(const __bf16* base, int ld) {
  const int l = laneId();
  const int n = l & 15;
  const int k0 = (l >> 4) << 4;
  const __bf16* p = base + n * ld + k0;
  union { v16bf v; uint4 q[2]; } b;
  b.q[0] = *(const uint4*)(p);
  b.q[1] = *(const uint4*)(p + 8);
  return b.v;
}

// f32 -> bf16 staging helpers.
union Cvt8 { __bf16 h[8]; uint4 u; };
union Cvt4 { __bf16 h[4]; uint2 u; };
static __device__ __forceinline__ void stage8_f32(__bf16* dst, const float* src) {
  const float4 f0 = *(const float4*)src;
  const float4 f1 = *(const float4*)(src + 4);
  Cvt8 c;
  c.h[0] = (__bf16)f0.x; c.h[1] = (__bf16)f0.y; c.h[2] = (__bf16)f0.z; c.h[3] = (__bf16)f0.w;
  c.h[4] = (__bf16)f1.x; c.h[5] = (__bf16)f1.y; c.h[6] = (__bf16)f1.z; c.h[7] = (__bf16)f1.w;
  *(uint4*)dst = c.u;
}

// ---- Kernel 1: fused Q/K/V projections: X @ W^T + b -> bf16 [B,NH,S,HD] ----
__global__ __launch_bounds__(256)
void k_qkv_proj(const float* __restrict__ Xq, const float* __restrict__ Xk,
                const float* __restrict__ Xv,
                const float* __restrict__ Wq, const float* __restrict__ bq,
                const float* __restrict__ Wk, const float* __restrict__ bk,
                const float* __restrict__ Wv, const float* __restrict__ bv,
                __bf16* __restrict__ Qo, __bf16* __restrict__ Ko,
                __bf16* __restrict__ Vo) {
  __shared__ __bf16 sA[64 * 32];
  __shared__ __bf16 sB[64 * 32];   // W tile, N-major (W is [n,k] row-major)
  const float *X, *W, *bias; __bf16* O;
  if (blockIdx.z == 0)      { X = Xq; W = Wq; bias = bq; O = Qo; }
  else if (blockIdx.z == 1) { X = Xk; W = Wk; bias = bk; O = Ko; }
  else                      { X = Xv; W = Wv; bias = bv; O = Vo; }

  const int n0 = blockIdx.x * 64;
  const int m0 = blockIdx.y * 64;
  const int t = threadIdx.x;
  const int srow = t >> 2, scol = (t & 3) * 8;
  const int w = waveId(), l = laneId();
  const int wm = w >> 1, wn = w & 1;            // 4x2 wave grid, 16x32 C/wave

  v8f acc0 = {}, acc1 = {};
  for (int kk = 0; kk < Hz; kk += 32) {
    stage8_f32(&sA[srow * 32 + scol], X + (size_t)(m0 + srow) * Hz + kk + scol);
    stage8_f32(&sB[srow * 32 + scol], W + (size_t)(n0 + srow) * Hz + kk + scol);
    __syncthreads();
    const v16bf a  = ldsLoadA(sA + wm * 16 * 32, 32);
    const v16bf b0 = ldsLoadB(sB + (wn * 32 + 0)  * 32, 32);
    const v16bf b1 = ldsLoadB(sB + (wn * 32 + 16) * 32, 32);
    acc0 = wmma_bf16(a, b0, acc0);
    acc1 = wmma_bf16(a, b1, acc1);
    __syncthreads();
  }
  // C/D: row = r + 8*(lane>=16), col = lane&15. Store head-split bf16.
  const int mbase = m0 + wm * 16 + (l >> 4) * 8;
  const int ncol0 = n0 + wn * 32 + (l & 15);
#pragma unroll
  for (int r = 0; r < 8; ++r) {
    const int m = mbase + r, bb = m >> 9, s = m & (Sz - 1);
    {
      const int n = ncol0, h = n >> 6, d = n & 63;
      O[(size_t)((bb * NHz + h) * Sz + s) * HDz + d] = (__bf16)(acc0[r] + bias[n]);
    }
    {
      const int n = ncol0 + 16, h = n >> 6, d = n & 63;
      O[(size_t)((bb * NHz + h) * Sz + s) * HDz + d] = (__bf16)(acc1[r] + bias[n]);
    }
  }
}

// ---- Kernel 2: per-head QK^T -> f32 scores (async bf16 tile staging) -------
__global__ __launch_bounds__(256)
void k_qk(const __bf16* __restrict__ Q, const __bf16* __restrict__ K,
          float* __restrict__ scores) {
  __shared__ __bf16 sA[64 * 32];
  __shared__ __bf16 sB[64 * 32];   // K tile: [keypos][d] == N-major for B=K^T
  const int bh = blockIdx.z;
  const int q0 = blockIdx.y * 64;
  const int k0 = blockIdx.x * 64;
  const __bf16* Qb = Q + (size_t)bh * Sz * HDz;
  const __bf16* Kb = K + (size_t)bh * Sz * HDz;
  const int t = threadIdx.x, srow = t >> 2, scol = (t & 3) * 8;
  const int w = waveId(), l = laneId();
  const int wm = w >> 1, wn = w & 1;
  v8f acc0 = {}, acc1 = {};
  for (int d0 = 0; d0 < HDz; d0 += 32) {
    asyncCopy16(&Qb[(size_t)(q0 + srow) * HDz + d0 + scol], &sA[srow * 32 + scol]);
    asyncCopy16(&Kb[(size_t)(k0 + srow) * HDz + d0 + scol], &sB[srow * 32 + scol]);
    asyncWait0();
    __syncthreads();
    const v16bf a  = ldsLoadA(sA + wm * 16 * 32, 32);
    const v16bf b0 = ldsLoadB(sB + (wn * 32 + 0)  * 32, 32);
    const v16bf b1 = ldsLoadB(sB + (wn * 32 + 16) * 32, 32);
    acc0 = wmma_bf16(a, b0, acc0);
    acc1 = wmma_bf16(a, b1, acc1);
    __syncthreads();
  }
  const int qrow = q0 + wm * 16 + (l >> 4) * 8;
  const int kc = k0 + wn * 32 + (l & 15);
  float* sp = scores + (size_t)bh * Sz * Sz;
#pragma unroll
  for (int r = 0; r < 8; ++r) {
    sp[(size_t)(qrow + r) * Sz + kc]      = acc0[r];
    sp[(size_t)(qrow + r) * Sz + kc + 16] = acc1[r];
  }
}

// ---- Kernel 3: scores += Q_heads . rel_k^T, per (b, query i): M=16 GEMM ----
__global__ __launch_bounds__(256)
void k_qr(const __bf16* __restrict__ Q, const float* __restrict__ relk,
          float* __restrict__ scores) {
  __shared__ __bf16 sA[16 * 64];     // heads x d
  __shared__ __bf16 sB[256 * 64];    // keypos x d (half of S), N-major: 32 KB
  const int bi = blockIdx.x;
  const int b = bi >> 9, i = bi & (Sz - 1);
  const int t = threadIdx.x;
  const int w = waveId(), l = laneId();
  {
    const int h = t >> 4, d = (t & 15) * 4;
    *(uint2*)&sA[h * 64 + d] =
        *(const uint2*)&Q[(size_t)((b * NHz + h) * Sz + i) * HDz + d];
  }
  const float* relbase = relk + (size_t)(b * Sz + i) * Sz * HDz;
  for (int half = 0; half < 2; ++half) {
    __syncthreads();
    const float* rh = relbase + (size_t)half * 256 * HDz;
    if (half == 0) {  // prefetch the second half while converting the first
      const float* nx = relbase + (size_t)256 * HDz;
      __builtin_prefetch(nx + (size_t)t * 64, 0, 0);
      __builtin_prefetch(nx + (size_t)t * 64 + 32, 0, 0);
    }
#pragma unroll
    for (int it = 0; it < 16; ++it) {          // 4096 float4s, coalesced
      const int e4 = it * 256 + t;
      const int row = e4 >> 4;
      const int col = (e4 & 15) * 4;
      const float4 f = *(const float4*)&rh[(size_t)row * HDz + col];
      Cvt4 c; c.h[0] = (__bf16)f.x; c.h[1] = (__bf16)f.y;
              c.h[2] = (__bf16)f.z; c.h[3] = (__bf16)f.w;
      *(uint2*)&sB[row * 64 + col] = c.u;
    }
    __syncthreads();
    v8f acc0 = {}, acc1 = {};
#pragma unroll
    for (int d0 = 0; d0 < HDz; d0 += 32) {
      const v16bf a  = ldsLoadA(sA + d0, 64);
      const v16bf b0 = ldsLoadB(sB + (w * 32 + 0)  * 64 + d0, 64);
      const v16bf b1 = ldsLoadB(sB + (w * 32 + 16) * 64 + d0, 64);
      acc0 = wmma_bf16(a, b0, acc0);
      acc1 = wmma_bf16(a, b1, acc1);
    }
    const int h0 = (l >> 4) * 8;
    const int n = half * 256 + w * 32 + (l & 15);
#pragma unroll
    for (int r = 0; r < 8; ++r) {
      const size_t idx = (size_t)((b * NHz + h0 + r) * Sz + i) * Sz + n;
      scores[idx]      += acc0[r];
      scores[idx + 16] += acc1[r];
    }
  }
}

// ---- Kernel 4: masked softmax, wave per row, write bf16 P ------------------
__global__ __launch_bounds__(256)
void k_softmax(const float* __restrict__ scores, const int* __restrict__ mask,
               __bf16* __restrict__ P) {
  const int w = waveId(), l = laneId();
  const int row = blockIdx.x * 8 + w;          // [0, B*NH*S)
  const int q = row & (Sz - 1);
  const int bh = row >> 9;
  const int b = bh >> 4;
  const float* src = scores + (size_t)row * Sz;
  const int* mrow = mask + (size_t)(b * Sz + q) * Sz;
  float v[16];
  float mx = -3.0e38f;
#pragma unroll
  for (int j = 0; j < 16; ++j) {
    const int k = j * 32 + l;
    float s = src[k] * 0.125f;                 // 1/sqrt(HD)
    if (mrow[k] == 0) s = -1000.0f;
    v[j] = s;
    mx = fmaxf(mx, s);
  }
#pragma unroll
  for (int o = 16; o > 0; o >>= 1) mx = fmaxf(mx, __shfl_xor(mx, o, 32));
  float sum = 0.f;
#pragma unroll
  for (int j = 0; j < 16; ++j) { v[j] = __expf(v[j] - mx); sum += v[j]; }
#pragma unroll
  for (int o = 16; o > 0; o >>= 1) sum += __shfl_xor(sum, o, 32);
  const float inv = 1.0f / sum;
  __bf16* dst = P + (size_t)row * Sz;
#pragma unroll
  for (int j = 0; j < 16; ++j) dst[j * 32 + l] = (__bf16)(v[j] * inv);
}

// ---- Kernel 5: per-head P @ V -> f32 attn [B,S,H] --------------------------
__global__ __launch_bounds__(256)
void k_wv(const __bf16* __restrict__ P, const __bf16* __restrict__ V,
          float* __restrict__ attn) {
  __shared__ __bf16 sA[64 * 32];
  __shared__ __bf16 sB[64 * 32];   // d x k (transposed during staging)
  const int bh = blockIdx.y;
  const int q0 = blockIdx.x * 64;
  const int b = bh >> 4, h = bh & 15;
  const __bf16* Pb = P + (size_t)bh * Sz * Sz;
  const __bf16* Vb = V + (size_t)bh * Sz * HDz;
  const int t = threadIdx.x, w = waveId(), l = laneId();
  const int wm = w >> 1, wn = w & 1;
  v8f acc0 = {}, acc1 = {};
  for (int k0 = 0; k0 < Sz; k0 += 32) {
    {   // A: async bf16 copy of the P tile
      const int srow = t >> 2, scol = (t & 3) * 8;
      asyncCopy16(&Pb[(size_t)(q0 + srow) * Sz + k0 + scol],
                  &sA[srow * 32 + scol]);
    }
    {   // B: transpose V during staging
      const int d = t >> 2, c0 = (t & 3) * 8;
#pragma unroll
      for (int cc = 0; cc < 8; ++cc)
        sB[d * 32 + c0 + cc] = Vb[(size_t)(k0 + c0 + cc) * HDz + d];
    }
    asyncWait0();
    __syncthreads();
    const v16bf a  = ldsLoadA(sA + wm * 16 * 32, 32);
    const v16bf b0 = ldsLoadB(sB + (wn * 32 + 0)  * 32, 32);
    const v16bf b1 = ldsLoadB(sB + (wn * 32 + 16) * 32, 32);
    acc0 = wmma_bf16(a, b0, acc0);
    acc1 = wmma_bf16(a, b1, acc1);
    __syncthreads();
  }
  const int q = q0 + wm * 16 + (l >> 4) * 8;
  const int d = wn * 32 + (l & 15);
#pragma unroll
  for (int r = 0; r < 8; ++r) {
    const size_t idx = (size_t)(b * Sz + q + r) * Hz + h * HDz + d;
    attn[idx]      = acc0[r];
    attn[idx + 16] = acc1[r];
  }
}

// ---- Kernel 6: attn += P_heads @ rel_v, per (b,i): M=16 GEMM, wave/i -------
__global__ __launch_bounds__(256)
void k_wr(const __bf16* __restrict__ P, const float* __restrict__ relv,
          float* __restrict__ attn) {
  __shared__ __bf16 sA[8][16 * 32];  //  8 KB, per-wave A staging
  __shared__ __bf16 sB[8][64 * 32];  // 32 KB, per-wave B staging (N-major)
  const int w = waveId(), l = laneId();
  const int bi = blockIdx.x * 8 + w;
  const int b = bi >> 9, i = bi & (Sz - 1);
  __bf16* aw = sA[w];
  __bf16* bw = sB[w];
  const float* relbase = relv + (size_t)(b * Sz + i) * Sz * HDz;
  v8f acc[4] = {};
  for (int k0 = 0; k0 < Sz; k0 += 32) {
    {   // A: async bf16 copy; each lane stages half a head row
      const int h = l >> 1, c0 = (l & 1) * 16;
      const __bf16* src = P + (size_t)((b * NHz + h) * Sz + i) * Sz + k0 + c0;
      asyncCopy16(src, &aw[h * 32 + c0]);
      asyncCopy16(src + 8, &aw[h * 32 + c0 + 8]);
    }
    if (k0 + 32 < Sz) {  // prefetch next rel_v block (32 rows x 256 B)
      const float* nx = &relbase[(size_t)(k0 + 32 + l) * HDz];
      __builtin_prefetch(nx, 0, 0);
      __builtin_prefetch(nx + 32, 0, 0);
    }
    {   // B: bw[d*32+c] = rel_v[b,i,k0+c,d]; coalesced float2 per lane
      const int d = l * 2;
#pragma unroll
      for (int cc = 0; cc < 32; ++cc) {
        const float2 f = *(const float2*)&relbase[(size_t)(k0 + cc) * HDz + d];
        bw[d * 32 + cc]       = (__bf16)f.x;
        bw[(d + 1) * 32 + cc] = (__bf16)f.y;
      }
    }
    // intra-wave producer->consumer: wait for async copies + DS stores.
    asm volatile("s_wait_asynccnt 0\n\ts_wait_dscnt 0" ::: "memory");
    const v16bf a = ldsLoadA(aw, 32);
#pragma unroll
    for (int j = 0; j < 4; ++j) {
      const v16bf bf = ldsLoadB(bw + (j * 16) * 32, 32);
      acc[j] = wmma_bf16(a, bf, acc[j]);
    }
  }
  const int h0 = (l >> 4) * 8;
  const int nl = l & 15;
#pragma unroll
  for (int j = 0; j < 4; ++j)
#pragma unroll
    for (int r = 0; r < 8; ++r) {
      const size_t idx =
          (size_t)(b * Sz + i) * Hz + (h0 + r) * HDz + j * 16 + nl;
      attn[idx] += acc[j][r];
    }
}

// ---- Kernel 7: output projection: attn @ Wo^T + bo -> f32 out --------------
__global__ __launch_bounds__(256)
void k_oproj(const float* __restrict__ Xin, const float* __restrict__ Wo,
             const float* __restrict__ bo, float* __restrict__ Out) {
  __shared__ __bf16 sA[64 * 32];
  __shared__ __bf16 sB[64 * 32];
  const int n0 = blockIdx.x * 64;
  const int m0 = blockIdx.y * 64;
  const int t = threadIdx.x, srow = t >> 2, scol = (t & 3) * 8;
  const int w = waveId(), l = laneId();
  const int wm = w >> 1, wn = w & 1;
  v8f acc0 = {}, acc1 = {};
  for (int kk = 0; kk < Hz; kk += 32) {
    stage8_f32(&sA[srow * 32 + scol], Xin + (size_t)(m0 + srow) * Hz + kk + scol);
    stage8_f32(&sB[srow * 32 + scol], Wo  + (size_t)(n0 + srow) * Hz + kk + scol);
    __syncthreads();
    const v16bf a  = ldsLoadA(sA + wm * 16 * 32, 32);
    const v16bf b0 = ldsLoadB(sB + (wn * 32 + 0)  * 32, 32);
    const v16bf b1 = ldsLoadB(sB + (wn * 32 + 16) * 32, 32);
    acc0 = wmma_bf16(a, b0, acc0);
    acc1 = wmma_bf16(a, b1, acc1);
    __syncthreads();
  }
  const int m = m0 + wm * 16 + (l >> 4) * 8;
  const int n = n0 + wn * 32 + (l & 15);
#pragma unroll
  for (int r = 0; r < 8; ++r) {
    Out[(size_t)(m + r) * Hz + n]      = acc0[r] + bo[n];
    Out[(size_t)(m + r) * Hz + n + 16] = acc1[r] + bo[n + 16];
  }
}

// ---- Launch ----------------------------------------------------------------
extern "C" void kernel_launch(void* const* d_in, const int* in_sizes, int n_in,
                              void* d_out, int out_size, void* d_ws, size_t ws_size,
                              hipStream_t stream) {
  (void)in_sizes; (void)n_in; (void)out_size; (void)ws_size;
  const float* query = (const float*)d_in[0];
  const float* key   = (const float*)d_in[1];
  const float* value = (const float*)d_in[2];
  const float* relk  = (const float*)d_in[3];
  const float* relv  = (const float*)d_in[4];
  const int*   mask  = (const int*)d_in[5];
  const float* Wq = (const float*)d_in[6];  const float* bq = (const float*)d_in[7];
  const float* Wk = (const float*)d_in[8];  const float* bk = (const float*)d_in[9];
  const float* Wv = (const float*)d_in[10]; const float* bv = (const float*)d_in[11];
  const float* Wo = (const float*)d_in[12]; const float* bo = (const float*)d_in[13];
  float* out = (float*)d_out;

  char* ws = (char*)d_ws;
  size_t off = 0;
  auto carve = [&](size_t bytes) -> char* {
    char* p = ws + off; off += (bytes + 255) & ~(size_t)255; return p;
  };
  __bf16* Qbf = (__bf16*)carve((size_t)Bz * Sz * Hz * 2);          //  4 MB
  __bf16* Kbf = (__bf16*)carve((size_t)Bz * Sz * Hz * 2);          //  4 MB
  __bf16* Vbf = (__bf16*)carve((size_t)Bz * Sz * Hz * 2);          //  4 MB
  float*  scores = (float*)carve((size_t)Bz * NHz * Sz * Sz * 4);  // 64 MB
  __bf16* P = (__bf16*)carve((size_t)Bz * NHz * Sz * Sz * 2);      // 32 MB
  float*  attn = (float*)carve((size_t)Bz * Sz * Hz * 4);          //  8 MB

  const dim3 blk(256);
  k_qkv_proj<<<dim3(Hz / 64, (Bz * Sz) / 64, 3), blk, 0, stream>>>(
      query, key, value, Wq, bq, Wk, bk, Wv, bv, Qbf, Kbf, Vbf);
  k_qk<<<dim3(Sz / 64, Sz / 64, Bz * NHz), blk, 0, stream>>>(Qbf, Kbf, scores);
  k_qr<<<dim3(Bz * Sz), blk, 0, stream>>>(Qbf, relk, scores);
  k_softmax<<<dim3(Bz * NHz * Sz / 8), blk, 0, stream>>>(scores, mask, P);
  k_wv<<<dim3(Sz / 64, Bz * NHz), blk, 0, stream>>>(P, Vbf, attn);
  k_wr<<<dim3(Bz * Sz / 8), blk, 0, stream>>>(P, relv, attn);
  k_oproj<<<dim3(Hz / 64, (Bz * Sz) / 64), blk, 0, stream>>>(attn, Wo, bo, out);
}
